// SetAbstraction_49211735277939
// MI455X (gfx1250) — matrix operand order repaired and checked by reading, not compiled, not run on hardware
//
#include <hip/hip_runtime.h>

typedef _Float16 half_t;
typedef __attribute__((ext_vector_type(16))) _Float16 v16h;
typedef __attribute__((ext_vector_type(8)))  float    v8f;

#define BB      8
#define NPTS    8192
#define MS      2048      // 0.25 * N
#define KNB     32
#define DIN     13
#define H1      64
#define H2      128
#define RAD2    0.04f     // R^2 = 0.2^2
#define SHSTR   48        // halves per lane-row in activation stash (96 B, 32B-aligned)

// ---------------------------------------------------------------------------
// 1) Farthest point sampling: one block per cloud, point coords + running min
//    distances live in registers (8 pts/thread), two-level argmax per step.
// ---------------------------------------------------------------------------
__global__ __launch_bounds__(1024) void fps_kernel(const float* __restrict__ pos,
                                                   int* __restrict__ idx) {
  const int b   = blockIdx.x;
  const int tid = threadIdx.x;
  const int lane = tid & 31;
  const int wid  = tid >> 5;
  const float* P = pos + (size_t)b * NPTS * 3;

  float px[8], py[8], pz[8], md[8];
#pragma unroll
  for (int t = 0; t < 8; ++t) {
    const int i = tid + t * 1024;
    px[t] = P[i * 3 + 0];
    py[t] = P[i * 3 + 1];
    pz[t] = P[i * 3 + 2];
    md[t] = 1e10f;
  }

  __shared__ float swv[32];
  __shared__ int   swi[32];
  __shared__ int   s_last;
  if (tid == 0) { idx[(size_t)b * MS] = 0; s_last = 0; }
  __syncthreads();

  for (int step = 1; step < MS; ++step) {
    const int last = s_last;
    const float lx = P[last * 3 + 0];
    const float ly = P[last * 3 + 1];
    const float lz = P[last * 3 + 2];

    float bv = -1.0f;
    int   bi = NPTS;
#pragma unroll
    for (int t = 0; t < 8; ++t) {
      const float dx = px[t] - lx, dy = py[t] - ly, dz = pz[t] - lz;
      const float d2 = dx * dx + dy * dy + dz * dz;
      const float nm = fminf(md[t], d2);
      md[t] = nm;
      const int i = tid + t * 1024;
      if (nm > bv || (nm == bv && i < bi)) { bv = nm; bi = i; }
    }
#pragma unroll
    for (int off = 16; off > 0; off >>= 1) {
      const float ov = __shfl_xor(bv, off, 32);
      const int   oi = __shfl_xor(bi, off, 32);
      if (ov > bv || (ov == bv && oi < bi)) { bv = ov; bi = oi; }
    }
    if (lane == 0) { swv[wid] = bv; swi[wid] = bi; }
    __syncthreads();
    if (wid == 0) {
      bv = swv[lane];
      bi = swi[lane];
#pragma unroll
      for (int off = 16; off > 0; off >>= 1) {
        const float ov = __shfl_xor(bv, off, 32);
        const int   oi = __shfl_xor(bi, off, 32);
        if (ov > bv || (ov == bv && oi < bi)) { bv = ov; bi = oi; }
      }
      if (lane == 0) { idx[(size_t)b * MS + step] = bi; s_last = bi; }
    }
    __syncthreads();
  }
}

// ---------------------------------------------------------------------------
// 2) Radius ball query, first K in-ball by index: one wave per center.
// ---------------------------------------------------------------------------
__global__ __launch_bounds__(256) void ball_query_kernel(const float* __restrict__ pos,
                                                         const int* __restrict__ idx,
                                                         int* __restrict__ nbr) {
  const int wid  = threadIdx.x >> 5;
  const int lane = threadIdx.x & 31;
  const int gid  = blockIdx.x * 8 + wid;          // center id in [0, B*MS)
  const int b    = gid / MS;
  const int ci   = idx[gid];
  const float* P = pos + (size_t)b * NPTS * 3;
  const float cx = P[ci * 3 + 0];
  const float cy = P[ci * 3 + 1];
  const float cz = P[ci * 3 + 2];

  int cnt = 0;
  for (int base = 0; base < NPTS && cnt < KNB; base += 32) {
    const int j = base + lane;
    const float dx = P[j * 3 + 0] - cx;
    const float dy = P[j * 3 + 1] - cy;
    const float dz = P[j * 3 + 2] - cz;
    const bool  in = (dx * dx + dy * dy + dz * dz) <= RAD2;
    const unsigned mask = (unsigned)__ballot(in);
    if (in) {
      const int off = cnt + __popc(mask & ((1u << lane) - 1u));
      if (off < KNB) nbr[(size_t)gid * KNB + off] = j;
    }
    cnt += __popc(mask);
  }
  if (lane >= cnt) nbr[(size_t)gid * KNB + lane] = -1;   // K == 32 == wave size
}

// ---------------------------------------------------------------------------
// 3) Gather -> MLP (WMMA f16, f32 accum) -> masked max over K.
//    Weights are staged into LDS already in WMMA B-operand register order so
//    each B-tile fetch is one aligned v16h (2 x ds_load_b128). Layer-1
//    activations are stashed pre-swizzled into A-operand order the same way.
// ---------------------------------------------------------------------------
__global__ __launch_bounds__(256) void sa_mlp_kernel(
    const float* __restrict__ x, const float* __restrict__ pos,
    const int* __restrict__ idx, const int* __restrict__ nbr,
    const float* __restrict__ W1, const float* __restrict__ b1,
    const float* __restrict__ W2, const float* __restrict__ b2,
    float* __restrict__ x_out) {
  __shared__ __align__(32) half_t sB1[4 * 32 * 16];    // [n][lane][h]
  __shared__ __align__(32) half_t sB2[16 * 32 * 16];   // [s*8+n][lane][h]
  __shared__ __align__(32) half_t sh[8 * 32 * SHSTR];  // per-wave, [lane][32 halves]
  __shared__ float sb1[H1];
  __shared__ float sb2[H2];

  // Stage W1 (zero-padded K 16->32) and W2 into B-operand register order.
  for (int e = threadIdx.x; e < 4 * 32 * 16; e += 256) {
    const int n = e >> 9, ln = (e >> 4) & 31, h = e & 15;
    const int khi = ln >> 4, kcol = ln & 15;
    const int k = ((h < 8) ? h : h + 8) + 8 * khi;    // B layout: K = (h<8 ? h : h+8) + 8*(lane>=16)
    sB1[e] = (k < 16) ? (half_t)W1[k * H1 + n * 16 + kcol] : (half_t)0.0f;
  }
  for (int e = threadIdx.x; e < 16 * 32 * 16; e += 256) {
    const int tile = e >> 9, ln = (e >> 4) & 31, h = e & 15;
    const int s = tile >> 3, n = tile & 7;
    const int khi = ln >> 4, kcol = ln & 15;
    const int k = 32 * s + ((h < 8) ? h : h + 8) + 8 * khi;
    sB2[e] = (half_t)W2[k * H2 + n * 16 + kcol];
  }
  if (threadIdx.x < H1) sb1[threadIdx.x] = b1[threadIdx.x];
  if (threadIdx.x < H2) sb2[threadIdx.x] = b2[threadIdx.x];
  __syncthreads();

  const int wid  = threadIdx.x >> 5;
  const int lane = threadIdx.x & 31;
  const int col  = lane & 15;
  const int hi   = lane >> 4;

  const v16h* vB1 = (const v16h*)sB1;
  const v16h* vB2 = (const v16h*)sB2;
  half_t* myh = sh + wid * 32 * SHSTR;

  const int gid = blockIdx.x * 8 + wid;            // center id
  const int b   = gid / MS;
  const int ci  = idx[gid];
  const float cx = pos[((size_t)b * NPTS + ci) * 3 + 0];
  const float cy = pos[((size_t)b * NPTS + ci) * 3 + 1];
  const float cz = pos[((size_t)b * NPTS + ci) * 3 + 2];

  float cmax[8];
#pragma unroll
  for (int n = 0; n < 8; ++n) cmax[n] = -__builtin_inff();

  for (int t = 0; t < 2; ++t) {
    const int krow = t * 16 + col;                 // neighbor slot this lane gathers
    const int nj   = nbr[(size_t)gid * KNB + krow];
    const unsigned vmask = (unsigned)__ballot(nj >= 0) & 0xFFFFu;  // valid rows 0..15

    // Build A tile (16 rows x 32 K, feat in K=0..15, zeros above).
    v16h A;
    if (nj >= 0) {
      const float* xr = x   + ((size_t)b * NPTS + nj) * DIN;
      const float* pr = pos + ((size_t)b * NPTS + nj) * 3;
      const float d0 = pr[0] - cx, d1 = pr[1] - cy, d2 = pr[2] - cz;
#pragma unroll
      for (int h = 0; h < 16; ++h) {
        const int c = ((h < 8) ? h : h + 8) + 8 * hi;   // A layout K swizzle
        float fv;
        if (c < DIN)       fv = xr[c];
        else if (c == 13)  fv = d0;
        else if (c == 14)  fv = d1;
        else if (c == 15)  fv = d2;
        else               fv = 0.0f;
        A[h] = (half_t)fv;
      }
    } else {
#pragma unroll
      for (int h = 0; h < 16; ++h) A[h] = (half_t)0.0f;
    }

    // Layer 1: relu(A @ W1 + b1); stash into LDS pre-swizzled to A-operand
    // order for layer 2: reader lane L reads [L][s*16+h] = act(row=L&15,
    // chan=32s + kmapA(h, L>=16)).
#pragma unroll
    for (int n = 0; n < 4; ++n) {
      v8f c0 = {};
      c0 = __builtin_amdgcn_wmma_f32_16x16x32_f16(false, A, false, vB1[n * 32 + lane],
                                                  (short)0, c0, false, false);
      const float bb = sb1[n * 16 + col];
      // inverse A-swizzle for this lane's output channel ch = n*16+col
      const int ch   = n * 16 + col;
      const int s    = ch >> 5;
      const int c32  = ch & 31;
      const int hi_r = (c32 >> 3) & 1;
      const int hh   = c32 - 8 * ((c32 >= 8) + (c32 >= 24));
      half_t* dst = myh + (hi_r * 16) * SHSTR + (s * 16 + hh);
#pragma unroll
      for (int r = 0; r < 8; ++r)
        dst[(r + 8 * hi) * SHSTR] = (half_t)fmaxf(c0[r] + bb, 0.0f);
    }

    // Two aligned v16h loads give both k-step A-tiles for layer 2.
    const v16h* myv = (const v16h*)(myh + lane * SHSTR);
    v16h A2[2];
    A2[0] = myv[0];
    A2[1] = myv[1];

    // Layer 2: [16x64] @ [64x128], K split into 2 WMMA k-steps, 8 n-tiles.
    v8f acc[8];
#pragma unroll
    for (int n = 0; n < 8; ++n) acc[n] = (v8f){};
#pragma unroll
    for (int s = 0; s < 2; ++s) {
#pragma unroll
      for (int n = 0; n < 8; ++n) {
        acc[n] = __builtin_amdgcn_wmma_f32_16x16x32_f16(
            false, A2[s], false, vB2[(s * 8 + n) * 32 + lane],
            (short)0, acc[n], false, false);
      }
    }

    // Bias + relu + masked max over this tile's 16 neighbor rows.
    const int rbase = 8 * hi;
#pragma unroll
    for (int n = 0; n < 8; ++n) {
      const float bb = sb2[n * 16 + col];
      float mx = -__builtin_inff();
#pragma unroll
      for (int r = 0; r < 8; ++r) {
        if ((vmask >> (r + rbase)) & 1u)
          mx = fmaxf(mx, fmaxf(acc[n][r] + bb, 0.0f));
      }
      cmax[n] = fmaxf(cmax[n], mx);
    }
  }

  // Fold the two lane halves (rows 0-7 vs 8-15) and store 128 channels.
#pragma unroll
  for (int n = 0; n < 8; ++n)
    cmax[n] = fmaxf(cmax[n], __shfl_xor(cmax[n], 16, 32));
  if (hi == 0) {
#pragma unroll
    for (int n = 0; n < 8; ++n)
      x_out[(size_t)gid * H2 + n * 16 + col] = cmax[n];
  }
}

// ---------------------------------------------------------------------------
// 4) pos_s and batch_s outputs.
// ---------------------------------------------------------------------------
__global__ __launch_bounds__(256) void finalize_kernel(const float* __restrict__ pos,
                                                       const int* __restrict__ idx,
                                                       float* __restrict__ pos_s,
                                                       int* __restrict__ batch_s) {
  const int gid = blockIdx.x * 256 + threadIdx.x;
  if (gid >= BB * MS) return;
  const int b  = gid / MS;
  const int ci = idx[gid];
  const float* pr = pos + ((size_t)b * NPTS + ci) * 3;
  pos_s[gid * 3 + 0] = pr[0];
  pos_s[gid * 3 + 1] = pr[1];
  pos_s[gid * 3 + 2] = pr[2];
  batch_s[gid] = b;
}

extern "C" void kernel_launch(void* const* d_in, const int* in_sizes, int n_in,
                              void* d_out, int out_size, void* d_ws, size_t ws_size,
                              hipStream_t stream) {
  (void)in_sizes; (void)n_in; (void)out_size; (void)ws_size;
  const float* x   = (const float*)d_in[0];
  const float* pos = (const float*)d_in[1];
  // d_in[2] (batch) is implied by layout; unused.
  const float* W1  = (const float*)d_in[3];
  const float* b1  = (const float*)d_in[4];
  const float* W2  = (const float*)d_in[5];
  const float* b2  = (const float*)d_in[6];

  int* idx = (int*)d_ws;                                   // [B*MS]
  int* nbr = (int*)((char*)d_ws + (size_t)BB * MS * 4);    // [B*MS*K]

  float* x_out   = (float*)d_out;                          // [B*MS*128]
  float* pos_s   = x_out + (size_t)BB * MS * H2;           // [B*MS*3]
  int*   batch_s = (int*)(pos_s + (size_t)BB * MS * 3);    // [B*MS]

  fps_kernel<<<BB, 1024, 0, stream>>>(pos, idx);
  ball_query_kernel<<<(BB * MS) / 8, 256, 0, stream>>>(pos, idx, nbr);
  sa_mlp_kernel<<<(BB * MS) / 8, 256, 0, stream>>>(x, pos, idx, nbr, W1, b1, W2, b2, x_out);
  finalize_kernel<<<(BB * MS + 255) / 256, 256, 0, stream>>>(pos, idx, pos_s, batch_s);
}
